// SoftDTW_57612691308681
// MI455X (gfx1250) — compile-verified, matrix operand order checked by reading, not run on hardware
//
#include <hip/hip_runtime.h>

typedef __attribute__((ext_vector_type(16))) _Float16 v16h;
typedef __attribute__((ext_vector_type(8)))  _Float16 v8h;
typedef __attribute__((ext_vector_type(4)))  _Float16 v4h;
typedef __attribute__((ext_vector_type(8)))  float    v8f;

#define N_SEQ 4096
#define KDIM  64
#define INF_F 1e10f

// ---------------------------------------------------------------------------
// Kernel 1 (prep): one thread per row of x or y.
//   - row norm            sx[i] = sum_k v^2
//   - split-f16 encoding  h = f16(v), l = f16(v - h)   (~22-bit combined)
// xh/xl/yh/yl are 512 KB each -> L2 resident; conversion done ONCE per row
// instead of 256x inside the WMMA kernel.
// ---------------------------------------------------------------------------
__global__ __launch_bounds__(256) void sdtw_prep_kernel(
    const float* __restrict__ x, const float* __restrict__ y,
    _Float16* __restrict__ xh, _Float16* __restrict__ xl,
    _Float16* __restrict__ yh, _Float16* __restrict__ yl,
    float* __restrict__ sx, float* __restrict__ sy) {
  int i = blockIdx.x * blockDim.x + threadIdx.x;   // 0 .. 2*N_SEQ-1
  bool isx = (i < N_SEQ);
  int r = isx ? i : i - N_SEQ;
  const float* row = (isx ? x : y) + (size_t)r * KDIM;
  _Float16* ho = (isx ? xh : yh) + (size_t)r * KDIM;
  _Float16* lo = (isx ? xl : yl) + (size_t)r * KDIM;

  float s = 0.f;
#pragma unroll
  for (int k = 0; k < KDIM; k += 4) {
    float4 v = *(const float4*)(row + k);
    s += v.x * v.x + v.y * v.y + v.z * v.z + v.w * v.w;
    _Float16 h0 = (_Float16)v.x, h1 = (_Float16)v.y;
    _Float16 h2 = (_Float16)v.z, h3 = (_Float16)v.w;
    v4h hv = {h0, h1, h2, h3};
    v4h lv = {(_Float16)(v.x - (float)h0), (_Float16)(v.y - (float)h1),
              (_Float16)(v.z - (float)h2), (_Float16)(v.w - (float)h3)};
    *(v4h*)(ho + k) = hv;
    *(v4h*)(lo + k) = lv;
  }
  if (isx) sx[r] = s; else sy[r] = s;
}

// ---------------------------------------------------------------------------
// Kernel 2: D = sx[i] + sy[j] - 2 * x . y^T   via v_wmma_f32_16x16x32_f16
// acc += Ah*Bh + Ah*Bl + Al*Bh  (lo*lo dropped, ~2^-22 relative error).
// One wave32 per 16x16 output tile; pure b128 loads -> WMMA, no VALU convert.
// Store-bound: 64 MB output @ 23.3 TB/s ~ 3 us; D stays resident in 192MB L2.
// ---------------------------------------------------------------------------
__device__ inline v16h load16(const _Float16* __restrict__ p) {
  // A/B f16 operand layout: per lane, K = kb+{0..7} U kb+{16..23}
  v8h a = *(const v8h*)p;          // 16B -> global_load_b128
  v8h b = *(const v8h*)(p + 16);   // 16B -> global_load_b128
  return __builtin_shufflevector(a, b, 0, 1, 2, 3, 4, 5, 6, 7,
                                 8, 9, 10, 11, 12, 13, 14, 15);
}

__global__ __launch_bounds__(256) void sdtw_dist_kernel(
    const _Float16* __restrict__ xh, const _Float16* __restrict__ xl,
    const _Float16* __restrict__ yh, const _Float16* __restrict__ yl,
    const float* __restrict__ sx, const float* __restrict__ sy,
    float* __restrict__ Dm) {
  int wave = (int)((blockIdx.x * blockDim.x + threadIdx.x) >> 5);  // 0..65535
  int lane = threadIdx.x & 31;
  int tm   = wave >> 8;       // tile row (256 tiles of 16)
  int tn   = wave & 255;      // tile col
  int half = lane >> 4;       // lanes 0-15 vs 16-31
  int lrow = lane & 15;

  size_t aoff = (size_t)(tm * 16 + lrow) * KDIM;  // A row  (M index = lrow)
  size_t boff = (size_t)(tn * 16 + lrow) * KDIM;  // B col  (N index = lrow)

  v8f acc = {};
#pragma unroll
  for (int c = 0; c < 2; ++c) {               // K = 64 -> two K=32 chunks
    int kb = c * 32 + half * 8;
    v16h ah = load16(xh + aoff + kb);
    v16h al = load16(xl + aoff + kb);
    v16h bh = load16(yh + boff + kb);
    v16h bl = load16(yl + boff + kb);
    acc = __builtin_amdgcn_wmma_f32_16x16x32_f16(false, ah, false, bh,
                                                 (short)0, acc, false, false);
    acc = __builtin_amdgcn_wmma_f32_16x16x32_f16(false, ah, false, bl,
                                                 (short)0, acc, false, false);
    acc = __builtin_amdgcn_wmma_f32_16x16x32_f16(false, al, false, bh,
                                                 (short)0, acc, false, false);
  }

  // C/D layout: VGPR v -> M = v + 8*half, N = lane & 15
  int ncol  = tn * 16 + lrow;
  float syn = sy[ncol];
#pragma unroll
  for (int v = 0; v < 8; ++v) {
    int mrow = tm * 16 + v + half * 8;
    Dm[(size_t)mrow * N_SEQ + ncol] = sx[mrow] + syn - 2.0f * acc[v];
  }
}

// ---------------------------------------------------------------------------
// Kernel 3: soft-DTW wavefront DP, single 1024-thread workgroup pipeline.
// Thread t owns dp columns [4t+1 .. 4t+4]; at step s it processes D-row s-t.
// Rightmost value per step flows to thread t+1 via a 3-slot rotating LDS buf
// (left boundary = step s-1, diag boundary = step s-2). One barrier per step.
// ---------------------------------------------------------------------------
__device__ inline float softmin3(float a, float b, float c) {
  // gamma = 1:  min - log(sum exp(min - .))   (stabilized logsumexp)
  float m = fminf(a, fminf(b, c));
  float s = __expf(m - a) + __expf(m - b) + __expf(m - c);
  return m - __logf(s);
}

__global__ __launch_bounds__(1024) void sdtw_dp_kernel(
    const float* __restrict__ Dm, float* __restrict__ out) {
  __shared__ float buf[3][1024];
  const int t = threadIdx.x;
  buf[0][t] = INF_F; buf[1][t] = INF_F; buf[2][t] = INF_F;   // dp[0][*] = INF
  float up0 = INF_F, up1 = INF_F, up2 = INF_F, up3 = INF_F;  // prev dp row
  __syncthreads();

  const int STEPS = N_SEQ + 1024 - 1;  // 5119 pipeline steps
  for (int s = 0; s < STEPS; ++s) {
    int r = s - t;  // D-row processed by this thread this step
    float leftb, diagb;
    if (t == 0) {
      leftb = INF_F;                        // dp[r+1][0] = INF
      diagb = (r == 0) ? 0.0f : INF_F;      // dp[0][0]=0, dp[r>0][0]=INF
    } else {
      leftb = buf[(s + 2) % 3][t - 1];      // neighbor's value from step s-1
      diagb = buf[(s + 1) % 3][t - 1];      // neighbor's value from step s-2
    }

    float right = INF_F;                    // inactive threads publish INF
    if (r >= 0 && r < N_SEQ) {
      float4 d4 = *(const float4*)(Dm + (size_t)r * N_SEQ + t * 4);
      float c0 = d4.x + softmin3(up0, leftb, diagb);
      float c1 = d4.y + softmin3(up1, c0, up0);
      float c2 = d4.z + softmin3(up2, c1, up1);
      float c3 = d4.w + softmin3(up3, c2, up2);
      up0 = c0; up1 = c1; up2 = c2; up3 = c3;
      right = c3;
      if (t == 1023 && r == N_SEQ - 1) out[0] = c3;   // dp[4096][4096]
    }
    buf[s % 3][t] = right;
    __syncthreads();
  }
}

// ---------------------------------------------------------------------------
extern "C" void kernel_launch(void* const* d_in, const int* in_sizes, int n_in,
                              void* d_out, int out_size, void* d_ws, size_t ws_size,
                              hipStream_t stream) {
  (void)in_sizes; (void)n_in; (void)out_size; (void)ws_size;
  const float* x = (const float*)d_in[0];
  const float* y = (const float*)d_in[1];

  // Workspace layout:
  float* sx = (float*)d_ws;                         // 4096 f32
  float* sy = sx + N_SEQ;                           // 4096 f32
  _Float16* xh = (_Float16*)(sy + N_SEQ);           // 4096*64 f16 (512 KB)
  _Float16* xl = xh + (size_t)N_SEQ * KDIM;
  _Float16* yh = xl + (size_t)N_SEQ * KDIM;
  _Float16* yl = yh + (size_t)N_SEQ * KDIM;
  float* Dm = (float*)(yl + (size_t)N_SEQ * KDIM);  // 4096*4096 f32 = 64 MB

  sdtw_prep_kernel<<<(2 * N_SEQ) / 256, 256, 0, stream>>>(x, y, xh, xl, yh, yl,
                                                          sx, sy);
  sdtw_dist_kernel<<<(256 * 256) / 8, 256, 0, stream>>>(xh, xl, yh, yl,
                                                        sx, sy, Dm);
  sdtw_dp_kernel<<<1, 1024, 0, stream>>>(Dm, (float*)d_out);
}
